// MultiheadDiffAttn_28913719837139
// MI455X (gfx1250) — matrix operand order repaired and checked by reading, not compiled, not run on hardware
//
#include <hip/hip_runtime.h>
#include <hip/hip_bf16.h>
#include <math.h>

// ---------------------------------------------------------------------------
// Types for gfx1250 WMMA (wave32). v16bf A/B fragments, v8f accumulator.
// ---------------------------------------------------------------------------
typedef __attribute__((ext_vector_type(16))) __bf16 bf16x16;
typedef __attribute__((ext_vector_type(8)))  __bf16 bf16x8;
typedef __attribute__((ext_vector_type(8)))  float  f32x8;

union BF16Frag { bf16x16 v; bf16x8 h[2]; };

static __device__ __forceinline__ f32x8 wmma_bf16(bf16x16 a, bf16x16 b, f32x8 c) {
  // D = A(16x32 bf16) * B(32x16 bf16) + C(16x16 f32)
  return __builtin_amdgcn_wmma_f32_16x16x32_bf16(
      /*neg_a=*/false, a, /*neg_b=*/false, b,
      /*c_mod=*/(short)0, c, /*reuse_a=*/false, /*reuse_b=*/false);
}

static __device__ __forceinline__ __bf16 f2bf(float f) {
  union { float f; unsigned u; } a; a.f = f;
  unsigned r = (a.u + 0x7FFFu + ((a.u >> 16) & 1u)) >> 16;  // RNE
  union { unsigned short s; __bf16 b; } o; o.s = (unsigned short)r;
  return o.b;
}

// Optional CDNA5 async global->LDS path (compile-safe guard).
#if defined(__AMDGCN__) && __has_builtin(__builtin_amdgcn_global_load_async_to_lds_b32)
#define HAVE_ASYNC_LDS 1
#else
#define HAVE_ASYNC_LDS 0
#endif

static __device__ __forceinline__ void wait_async_zero() {
#if defined(__AMDGCN__)
#if __has_builtin(__builtin_amdgcn_s_wait_asynccnt)
  __builtin_amdgcn_s_wait_asynccnt(0);
#else
  asm volatile("s_wait_asynccnt 0" ::: "memory");
#endif
#endif
}

// ---------------------------------------------------------------------------
// Constants
// ---------------------------------------------------------------------------
#define T_SEQ   2048
#define E_DIM   2048
#define KV_DIM  1024
#define H_OUT   16      // output heads
#define HD      64      // per-phase head dim
#define CHUNK   256     // key chunk for online softmax (divides T_SEQ)

// ---------------------------------------------------------------------------
// Elementwise f32 -> bf16 convert
// ---------------------------------------------------------------------------
__global__ void convert_f32_bf16(const float* __restrict__ in,
                                 __bf16* __restrict__ out, int n) {
  int i = blockIdx.x * blockDim.x + threadIdx.x;
  if (i < n) out[i] = f2bf(in[i]);
}

// ---------------------------------------------------------------------------
// Tiled transpose + convert: in[R][C] f32 -> out[C][R] bf16
// (R, C are multiples of 32 for every launch in this file)
// ---------------------------------------------------------------------------
__global__ __launch_bounds__(256)
void transpose_f32_bf16(const float* __restrict__ in, __bf16* __restrict__ out,
                        int R, int C) {
  __shared__ float tile[32][33];
  const int tx = threadIdx.x & 31, ty = threadIdx.x >> 5;  // 32 x 8
  const int r0 = blockIdx.y * 32, c0 = blockIdx.x * 32;
#if HAVE_ASYNC_LDS
  for (int i = 0; i < 32; i += 8) {
    int r = r0 + ty + i, c = c0 + tx;
    __builtin_amdgcn_global_load_async_to_lds_b32(
        (int*)(in + (size_t)r * C + c),
        (int*)&tile[ty + i][tx],
        /*offset=*/0, /*cpol=*/0);
  }
  wait_async_zero();
#else
  for (int i = 0; i < 32; i += 8) {
    int r = r0 + ty + i, c = c0 + tx;
    tile[ty + i][tx] = (r < R && c < C) ? in[(size_t)r * C + c] : 0.f;
  }
#endif
  __syncthreads();
  for (int i = 0; i < 32; i += 8) {
    int c = c0 + ty + i, r = r0 + tx;
    if (c < C && r < R) out[(size_t)c * R + r] = f2bf(tile[tx][ty + i]);
  }
}

// ---------------------------------------------------------------------------
// Lambda scalar:  lam = exp(lq1.lk1) - exp(lq2.lk2) + 0.2
// ---------------------------------------------------------------------------
__global__ void lambda_kernel(const float* lq1, const float* lk1,
                              const float* lq2, const float* lk2, float* lam) {
  __shared__ float s1[64], s2[64];
  int t = threadIdx.x;
  s1[t] = lq1[t] * lk1[t];
  s2[t] = lq2[t] * lk2[t];
  __syncthreads();
  if (t == 0) {
    float a = 0.f, b = 0.f;
    for (int i = 0; i < 64; ++i) { a += s1[i]; b += s2[i]; }
    lam[0] = __expf(a) - __expf(b) + 0.2f;
  }
}

// ---------------------------------------------------------------------------
// WMMA GEMM:  C[M,N] = A[M,K](bf16, row-major) * Bt[N,K](bf16)^T  (+ bias)
// Outputs: optional f32 C, optional bf16 C (scaled, optionally transposed).
// Block: 256 threads = 8 waves (4x2), block tile 128x128,
// wave tile 32x64 = 2x4 WMMA tiles -> 8 wmma per 12 b128 loads per k-step.
// ---------------------------------------------------------------------------
__global__ __launch_bounds__(256)
void gemm_bf16_wmma(const __bf16* __restrict__ A, const __bf16* __restrict__ Bt,
                    const float* __restrict__ bias,
                    float* __restrict__ Cf, __bf16* __restrict__ Cb,
                    int M, int N, int K, float scale, int transOut) {
  const int lane = threadIdx.x & 31;
  const int w    = threadIdx.x >> 5;            // 0..7
  const int wr   = w >> 1, wc = w & 1;          // 4 x 2 wave grid
  const int row0 = blockIdx.y * 128 + wr * 32;  // wave rows [row0, row0+32)
  const int col0 = blockIdx.x * 128 + wc * 64;  // wave cols [col0, col0+64)

  const int l15  = lane & 15;
  const int kb8  = (lane >> 4) * 8;             // A half-wave K offset
  const int kb16 = (lane >> 4) * 16;            // unused (B loads use kb8 as well)

  const __bf16* Arow0 = A + (size_t)(row0 + l15) * K;
  const __bf16* Arow1 = A + (size_t)(row0 + 16 + l15) * K;
  const __bf16* Brow[4];
#pragma unroll
  for (int t = 0; t < 4; ++t) Brow[t] = Bt + (size_t)(col0 + t * 16 + l15) * K;

  f32x8 acc[2][4];
#pragma unroll
  for (int i = 0; i < 2; ++i)
#pragma unroll
    for (int t = 0; t < 4; ++t) acc[i][t] = (f32x8){};

  for (int k = 0; k < K; k += 32) {
    __builtin_prefetch(Arow0 + k + 512, 0, 1);
    __builtin_prefetch(Arow1 + k + 512, 0, 1);
    BF16Frag a0, a1;
    a0.h[0] = *(const bf16x8*)(Arow0 + k + kb8);
    a0.h[1] = *(const bf16x8*)(Arow0 + k + 16 + kb8);
    a1.h[0] = *(const bf16x8*)(Arow1 + k + kb8);
    a1.h[1] = *(const bf16x8*)(Arow1 + k + 16 + kb8);
#pragma unroll
    for (int t = 0; t < 4; ++t) {
      BF16Frag b;
      b.h[0] = *(const bf16x8*)(Brow[t] + k + kb8);
      b.h[1] = *(const bf16x8*)(Brow[t] + k + 16 + kb8);
      acc[0][t] = wmma_bf16(a0.v, b.v, acc[0][t]);
      acc[1][t] = wmma_bf16(a1.v, b.v, acc[1][t]);
    }
  }

  // Epilogue: C layout -> row = rowTile + 8*(lane>=16) + e, col = lane&15
  const int mB = (lane >> 4) * 8;
#pragma unroll
  for (int i = 0; i < 2; ++i) {
    const int mBase = row0 + i * 16 + mB;
#pragma unroll
    for (int t = 0; t < 4; ++t) {
      const int n = col0 + t * 16 + l15;
      const float bv = bias ? bias[n] : 0.f;
#pragma unroll
      for (int e = 0; e < 8; ++e) {
        const int mm = mBase + e;
        float v = acc[i][t][e] + bv;
        if (Cf) Cf[(size_t)mm * N + n] = v;
        if (Cb) {
          __bf16 ov = f2bf(v * scale);
          if (transOut) Cb[(size_t)n * M + mm] = ov;
          else          Cb[(size_t)mm * N + n] = ov;
        }
      }
    }
  }
}

// ---------------------------------------------------------------------------
// Differential attention (causal, GQA, online softmax, WMMA):
//   block = (query block of 16 rows, output head h); 4 waves / 128 threads.
//   qb : [T, 2048] bf16 (pre-scaled by hd^-0.5); q head g = 2h+p at cols g*64
//   kb : [T, 1024] bf16; k head h at cols h*64
//   vtb: [1024, T] bf16 (V transposed); v head h>>1 at rows (h>>1)*128
//   out: attnb [T, 2048] bf16 = rmsnorm(O0 - lam*O1) * g * (1-0.2)
// ---------------------------------------------------------------------------
__global__ __launch_bounds__(128)
void diffattn_kernel(const __bf16* __restrict__ qb, const __bf16* __restrict__ kb,
                     const __bf16* __restrict__ vtb, const float* __restrict__ lamBuf,
                     const float* __restrict__ gsub, __bf16* __restrict__ attnb) {
  __shared__ float  sC[16][CHUNK];        // raw scores chunk
  __shared__ __bf16 sP[16][CHUNK];        // probabilities chunk (bf16)
  __shared__ float  sO[2][16][128];       // per-phase output accumulators
  __shared__ float  sM[16], sL[16];       // online-softmax stats
  __shared__ float  sRed[16][8];          // per-row partial reductions

  const int qblk = blockIdx.x;            // 0..127
  const int h    = blockIdx.y;            // 0..15
  const int tid  = threadIdx.x;
  const int lane = tid & 31, w = tid >> 5;
  const int t0   = qblk * 16;
  const int nK   = t0 + 16;               // causal: keys 0..nK-1 relevant
  const int numChunks = (nK + CHUNK - 1) / CHUNK;
  const float lam = lamBuf[0];

  // row / sub-lane mapping for the 8-threads-per-row reduction phases
  const int r   = tid >> 3;               // 0..15
  const int sub = tid & 7;                // 0..7
  const int trow = t0 + r;

  for (int i = tid; i < 2 * 16 * 128; i += 128) ((float*)sO)[i] = 0.f;
  __syncthreads();

  const int kb8  = (lane >> 4) * 8;
  const int kb16 = (lane >> 4) * 16;

  for (int p = 0; p < 2; ++p) {
    // ---- load Q fragments for this phase (constant across key chunks) ----
    const int ghead = 2 * h + p;
    const int mrow  = t0 + (lane & 15);
    const __bf16* qrow = qb + (size_t)mrow * E_DIM + ghead * HD;
    BF16Frag aq0, aq1;                    // d = 0..31 and d = 32..63
    aq0.h[0] = *(const bf16x8*)(qrow + 0  + kb8);
    aq0.h[1] = *(const bf16x8*)(qrow + 16 + kb8);
    aq1.h[0] = *(const bf16x8*)(qrow + 32 + kb8);
    aq1.h[1] = *(const bf16x8*)(qrow + 48 + kb8);

    if (tid < 16) { sM[tid] = -INFINITY; sL[tid] = 0.f; }
    __syncthreads();

    for (int c = 0; c < numChunks; ++c) {
      const int s0 = c * CHUNK;

      // ---- scores S = q . k^T for this chunk (16 tiles, 4 per wave) ----
      for (int tIdx = w; tIdx < CHUNK / 16; tIdx += 4) {
        const int sCol = s0 + tIdx * 16 + (lane & 15);
        const __bf16* krow = kb + (size_t)sCol * KV_DIM + h * HD;
        BF16Frag b0, b1;
        b0.h[0] = *(const bf16x8*)(krow + kb8);
        b0.h[1] = *(const bf16x8*)(krow + 16 + kb8);
        b1.h[0] = *(const bf16x8*)(krow + 32 + kb8);
        b1.h[1] = *(const bf16x8*)(krow + 48 + kb8);
        f32x8 sc = {};
        sc = wmma_bf16(aq0.v, b0.v, sc);
        sc = wmma_bf16(aq1.v, b1.v, sc);
        const int mB  = (lane >> 4) * 8;
        const int col = tIdx * 16 + (lane & 15);
        for (int i = 0; i < 8; ++i) sC[mB + i][col] = sc[i];
      }
      __syncthreads();

      // ---- online softmax update (8 threads per row) ----
      // 1) partial max over this thread's 32 columns
      float pm = -INFINITY;
      for (int j = sub * 32; j < sub * 32 + 32; ++j) {
        int s = s0 + j;
        if (s <= trow) pm = fmaxf(pm, sC[r][j]);
      }
      sRed[r][sub] = pm;
      __syncthreads();
      // 2) each thread reduces row max, computes rescale factor
      float mc = sRed[r][0];
      for (int i = 1; i < 8; ++i) mc = fmaxf(mc, sRed[r][i]);
      const float mOld = sM[r];
      const float mNew = fmaxf(mOld, mc);
      const float f    = __expf(mOld - mNew);  // 0 on first chunk (mOld=-inf)
      // 3) exponentiate + partial sum + write P; rescale O accumulator
      float ps = 0.f;
      for (int j = sub * 32; j < sub * 32 + 32; ++j) {
        int s = s0 + j;
        float pv = 0.f;
        if (s <= trow) { pv = __expf(sC[r][j] - mNew); ps += pv; }
        sP[r][j] = f2bf(pv);
      }
      for (int d = sub * 16; d < sub * 16 + 16; ++d) sO[p][r][d] *= f;
      __syncthreads();
      sRed[r][sub] = ps;
      __syncthreads();
      if (sub == 0) {
        float lsum = sL[r] * f;
        for (int i = 0; i < 8; ++i) lsum += sRed[r][i];
        sL[r] = lsum; sM[r] = mNew;
      }
      __syncthreads();

      // ---- O += P @ V for this chunk (8 d-tiles, 2 per wave) ----
      {
        const int d0  = w * 32;
        const int nD0 = d0 + (lane & 15), nD1 = nD0 + 16;
        const __bf16* v0row = vtb + (size_t)((h >> 1) * 128 + nD0) * T_SEQ;
        const __bf16* v1row = vtb + (size_t)((h >> 1) * 128 + nD1) * T_SEQ;
        const int mA = lane & 15;
        f32x8 o0 = {}, o1 = {};
        for (int ss = 0; ss < CHUNK / 32; ++ss) {
          BF16Frag a, b0, b1;
          a.h[0] = *(const bf16x8*)(&sP[mA][ss * 32 + kb16]);
          a.h[1] = *(const bf16x8*)(&sP[mA][ss * 32 + kb16 + 8]);
          const int sg = s0 + ss * 32 + kb16;
          b0.h[0] = *(const bf16x8*)(v0row + sg);
          b0.h[1] = *(const bf16x8*)(v0row + sg + 8);
          b1.h[0] = *(const bf16x8*)(v1row + sg);
          b1.h[1] = *(const bf16x8*)(v1row + sg + 8);
          o0 = wmma_bf16(a.v, b0.v, o0);
          o1 = wmma_bf16(a.v, b1.v, o1);
        }
        const int mB  = (lane >> 4) * 8;
        const int c0n = d0 + (lane & 15), c1n = c0n + 16;
        for (int i = 0; i < 8; ++i) {
          sO[p][mB + i][c0n] += o0[i];
          sO[p][mB + i][c1n] += o1[i];
        }
      }
      __syncthreads();
    }

    // ---- finalize phase: divide by softmax denominator (8 threads/row) ----
    {
      const float inv = 1.f / sL[r];
      for (int d = sub * 16; d < sub * 16 + 16; ++d) sO[p][r][d] *= inv;
    }
    __syncthreads();
  }

  // ---- combine phases, RMSNorm over 128 dims, scale, write bf16 ----
  {
    float psum = 0.f;
    for (int d = sub * 16; d < sub * 16 + 16; ++d) {
      float v = sO[0][r][d] - lam * sO[1][r][d];
      psum += v * v;
    }
    sRed[r][sub] = psum;
    __syncthreads();
    float ssum = 0.f;
    for (int i = 0; i < 8; ++i) ssum += sRed[r][i];
    const float rms = rsqrtf(ssum * (1.f / 128.f) + 1e-6f);
    const float outScale = 0.8f;          // (1 - LAMBDA_INIT)
    __bf16* orow = attnb + (size_t)trow * E_DIM + h * 128;
    for (int d = sub * 16; d < sub * 16 + 16; ++d) {
      float v = sO[0][r][d] - lam * sO[1][r][d];
      orow[d] = f2bf(v * rms * gsub[d] * outScale);
    }
  }
}

// ---------------------------------------------------------------------------
// Host-side launch
// ---------------------------------------------------------------------------
extern "C" void kernel_launch(void* const* d_in, const int* in_sizes, int n_in,
                              void* d_out, int out_size, void* d_ws, size_t ws_size,
                              hipStream_t stream) {
  (void)in_sizes; (void)n_in; (void)out_size; (void)ws_size;
  const float* x   = (const float*)d_in[0];
  const float* Wq  = (const float*)d_in[1];
  const float* bq  = (const float*)d_in[2];
  const float* Wk  = (const float*)d_in[3];
  const float* bk  = (const float*)d_in[4];
  const float* Wv  = (const float*)d_in[5];
  const float* bv  = (const float*)d_in[6];
  const float* Wo  = (const float*)d_in[7];
  const float* bo  = (const float*)d_in[8];
  const float* lq1 = (const float*)d_in[9];
  const float* lk1 = (const float*)d_in[10];
  const float* lq2 = (const float*)d_in[11];
  const float* lk2 = (const float*)d_in[12];
  const float* gsb = (const float*)d_in[13];
  float* out = (float*)d_out;

  // ---- workspace layout (bf16 buffers) ----
  char* ws = (char*)d_ws;
  size_t off = 0;
  auto alloc = [&](size_t bytes) { char* p = ws + off; off += (bytes + 255) & ~(size_t)255; return p; };
  __bf16* xb    = (__bf16*)alloc((size_t)T_SEQ * E_DIM * 2);   // x bf16
  __bf16* WqT   = (__bf16*)alloc((size_t)E_DIM * E_DIM * 2);   // Wq^T
  __bf16* WkT   = (__bf16*)alloc((size_t)KV_DIM * E_DIM * 2);  // Wk^T
  __bf16* WvT   = (__bf16*)alloc((size_t)KV_DIM * E_DIM * 2);  // Wv^T
  __bf16* WoT   = (__bf16*)alloc((size_t)E_DIM * E_DIM * 2);   // Wo^T
  __bf16* qb    = (__bf16*)alloc((size_t)T_SEQ * E_DIM * 2);   // scaled Q
  __bf16* kbuf  = (__bf16*)alloc((size_t)T_SEQ * KV_DIM * 2);  // K
  __bf16* vtb   = (__bf16*)alloc((size_t)KV_DIM * T_SEQ * 2);  // V^T
  __bf16* attnb = (__bf16*)alloc((size_t)T_SEQ * E_DIM * 2);   // attn activations
  float*  lam   = (float*)alloc(16);

  // 1) conversions
  {
    int n = T_SEQ * E_DIM;
    convert_f32_bf16<<<(n + 255) / 256, 256, 0, stream>>>(x, xb, n);
  }
  transpose_f32_bf16<<<dim3(E_DIM / 32, E_DIM / 32), 256, 0, stream>>>(Wq, WqT, E_DIM, E_DIM);
  transpose_f32_bf16<<<dim3(KV_DIM / 32, E_DIM / 32), 256, 0, stream>>>(Wk, WkT, E_DIM, KV_DIM);
  transpose_f32_bf16<<<dim3(KV_DIM / 32, E_DIM / 32), 256, 0, stream>>>(Wv, WvT, E_DIM, KV_DIM);
  transpose_f32_bf16<<<dim3(E_DIM / 32, E_DIM / 32), 256, 0, stream>>>(Wo, WoT, E_DIM, E_DIM);
  lambda_kernel<<<1, 64, 0, stream>>>(lq1, lk1, lq2, lk2, lam);

  // 2) projections (bf16 WMMA GEMMs, bf16 outputs). Q scaled by hd^-0.5.
  gemm_bf16_wmma<<<dim3(E_DIM / 128, T_SEQ / 128), 256, 0, stream>>>(
      xb, WqT, bq, nullptr, qb, T_SEQ, E_DIM, E_DIM, 0.125f, 0);
  gemm_bf16_wmma<<<dim3(KV_DIM / 128, T_SEQ / 128), 256, 0, stream>>>(
      xb, WkT, bk, nullptr, kbuf, T_SEQ, KV_DIM, E_DIM, 1.0f, 0);
  gemm_bf16_wmma<<<dim3(KV_DIM / 128, T_SEQ / 128), 256, 0, stream>>>(
      xb, WvT, bv, nullptr, vtb, T_SEQ, KV_DIM, E_DIM, 1.0f, /*transOut=*/1);

  // 3) differential attention + sub-LN
  diffattn_kernel<<<dim3(T_SEQ / 16, H_OUT), 128, 0, stream>>>(
      qb, kbuf, vtb, lam, gsb, attnb);

  // 4) output projection -> f32 d_out (+ bo)
  gemm_bf16_wmma<<<dim3(E_DIM / 128, T_SEQ / 128), 256, 0, stream>>>(
      attnb, WoT, bo, out, nullptr, T_SEQ, E_DIM, E_DIM, 1.0f, 0);
}
